// MedicalKnowledgeGraphModel_25477746000168
// MI455X (gfx1250) — compile-verified, matrix operand order checked by reading, not run on hardware
//
#include <hip/hip_runtime.h>
#include <hip/hip_bf16.h>
#include <stdint.h>

// ---------------------------------------------------------------------------
// MedicalKnowledgeGraphModel — hetero GraphSAGE on MI455X (gfx1250, wave32)
//
// Roofline: ~120 GFLOP of 128x128 GEMMs vs ~17 GB of random scatter/gather
// (5.6M edges x 3 layers x 1KB) -> aggregation-bound (~0.75 ms at 23.3 TB/s).
// GEMMs run on v_wmma_f32_16x16x32_bf16 with bf16-packed operands:
//   A tiles (activations) stream from global as 128-bit loads,
//   B tiles (weights) are staged into LDS once per block by the Tensor Data
//   Mover (tensor_load_to_lds + s_wait_tensorcnt) and read via ds_load_b128,
// so the inner loop is 2x global b128 + 2x ds b128 + 1x wmma per K=32 step.
// ---------------------------------------------------------------------------

typedef __attribute__((ext_vector_type(8)))  float        v8f;
typedef __attribute__((ext_vector_type(16))) __bf16       v16bf;
typedef __attribute__((ext_vector_type(4)))  unsigned int u32x4;
typedef __attribute__((ext_vector_type(8)))  int          i32x8;
typedef __attribute__((ext_vector_type(4)))  int          i32x4;

#define HID  128
#define OUTF 64

// ------------------------------ utility kernels ----------------------------

__global__ void k_zero(float* __restrict__ p, long long n) {
  long long t = (long long)blockIdx.x * blockDim.x + threadIdx.x;
  if (t < n) p[t] = 0.0f;
}

__global__ void k_copy(const float* __restrict__ s, float* __restrict__ d, long long n) {
  long long t = (long long)blockIdx.x * blockDim.x + threadIdx.x;
  if (t < n) d[t] = s[t];
}

// ------------------------------ encoder ------------------------------------

// out[n, j] = sum_k x[n,k] * W[j,k] + b[j]   (K = 32 or 48, tiny -> VALU)
__global__ void k_enc_linear(const float* __restrict__ x, const float* __restrict__ W,
                             const float* __restrict__ bias, float* __restrict__ out,
                             int nrows, int K) {
  long long t = (long long)blockIdx.x * blockDim.x + threadIdx.x;
  if (t >= (long long)nrows * HID) return;
  int n = (int)(t >> 7), j = (int)(t & 127);
  const float* xr = x + (size_t)n * K;
  const float* wr = W + (size_t)j * K;
  float s = bias[j];
  for (int k = 0; k < K; ++k) s += xr[k] * wr[k];
  out[(size_t)n * HID + j] = s;
}

// ------------------------------ batch-norm ---------------------------------

// stats[0:128] = column sums, stats[128:256] = column sum-of-squares
__global__ void k_bn_stats(const float* __restrict__ x, float* __restrict__ stats,
                           int nrows) {
  __shared__ float ssum[HID], ssq[HID];
  int tid = threadIdx.x;
  if (tid < HID) { ssum[tid] = 0.0f; ssq[tid] = 0.0f; }
  __syncthreads();
  int col = tid & 127;
  int rowsPerBlk = blockDim.x >> 7;                    // 2 rows / 256-thread block
  float s = 0.0f, q = 0.0f;
  for (int r = blockIdx.x * rowsPerBlk + (tid >> 7); r < nrows;
       r += gridDim.x * rowsPerBlk) {
    float v = x[(size_t)r * HID + col];
    s += v; q += v * v;
  }
  atomicAdd(&ssum[col], s);
  atomicAdd(&ssq[col], q);
  __syncthreads();
  if (tid < HID) {
    atomicAdd(&stats[tid], ssum[tid]);
    atomicAdd(&stats[HID + tid], ssq[tid]);
  }
}

__global__ void k_bn_apply(float* __restrict__ x, const float* __restrict__ stats,
                           const float* __restrict__ g, const float* __restrict__ b,
                           int nrows, int relu) {
  long long t = (long long)blockIdx.x * blockDim.x + threadIdx.x;
  if (t >= (long long)nrows * HID) return;
  int col = (int)(t & 127);
  float n   = (float)nrows;
  float mu  = stats[col] / n;
  float var = stats[HID + col] / n - mu * mu;          // biased variance (jnp.var)
  float v = (x[t] - mu) * rsqrtf(var + 1e-5f) * g[col] + b[col];
  if (relu) v = fmaxf(v, 0.0f);
  x[t] = v;
}

// ------------------------------ graph scatter ------------------------------

// segment_sum: per edge, 32 lanes x float4 atomic-add into msum[dst]; lane0
// bumps the count.  Random-access HBM traffic here is the workload bottleneck.
__global__ void k_scatter_add(const float* __restrict__ hsrc,
                              const int* __restrict__ src, const int* __restrict__ dst,
                              float* __restrict__ msum, float* __restrict__ cnt,
                              int E) {
  long long t = (long long)blockIdx.x * blockDim.x + threadIdx.x;
  int e = (int)(t >> 5);
  int l = (int)(t & 31);
  if (e >= E) return;
  int s = src[e], d = dst[e];
  float4 hv = *(const float4*)(hsrc + (size_t)s * HID + l * 4);  // global_load_b128
  float* ms = msum + (size_t)d * HID + l * 4;
  atomicAdd(ms + 0, hv.x);
  atomicAdd(ms + 1, hv.y);
  atomicAdd(ms + 2, hv.z);
  atomicAdd(ms + 3, hv.w);
  if (l == 0) atomicAdd(cnt + d, 1.0f);
}

// ------------------------------ bf16 operand packing -----------------------

// Apack[n][0:128]   = msum[n] / max(cnt[n],1)   (segment mean)
// Apack[n][128:256] = hdst[n]
__global__ void k_pack_a(const float* __restrict__ msum, const float* __restrict__ cnt,
                         const float* __restrict__ hdst,
                         unsigned short* __restrict__ apack_u16, int Nd) {
  long long t = (long long)blockIdx.x * blockDim.x + threadIdx.x;
  if (t >= (long long)Nd * 256) return;
  __bf16* apack = (__bf16*)apack_u16;
  int n = (int)(t >> 8), k = (int)(t & 255);
  float v;
  if (k < HID) v = msum[(size_t)n * HID + k] * (1.0f / fmaxf(cnt[n], 1.0f));
  else         v = hdst[(size_t)n * HID + (k - HID)];
  apack[t] = (__bf16)v;
}

// Bpack[n][0:128] = Wl[n][:],  Bpack[n][128:256] = Wr[n][:]   (row n = out col)
__global__ void k_pack_w(const float* __restrict__ Wl, const float* __restrict__ Wr,
                         unsigned short* __restrict__ bpack_u16) {
  int t = blockIdx.x * blockDim.x + threadIdx.x;
  if (t >= HID * 256) return;
  __bf16* bpack = (__bf16*)bpack_u16;
  int n = t >> 8, k = t & 255;
  float v = (k < HID) ? Wl[(size_t)n * HID + k] : Wr[(size_t)n * HID + (k - HID)];
  bpack[t] = (__bf16)v;
}

// ------------------------------ WMMA SAGE GEMM -----------------------------

// out[n, 0:128] = Apack[n] @ Bpack^T + bl   with Apack = [mean | h_dst],
// Bpack = [Wl | Wr].  One wave per 16x16 output tile; K=256 in 8 steps of
// v_wmma_f32_16x16x32_bf16.  combine==1 -> elementwise max into out.
//
// The block's 16x256 bf16 weight tile (8KB) is DMA'd into LDS once by the
// Tensor Data Mover; all 4 waves (4 row tiles) then read B from LDS.
//
// bf16 A 16x32 layout: lane L holds row M=L%16; half-slot e<8 -> K=e+8*(L/16),
// e>=8 -> K=8+e+8*(L/16).  Both runs are contiguous -> two 16B loads per step.
// B mirrored with N=L%16.  C/D: VGPR r -> M=r+8*(L/16), N=L%16.
__global__ void __launch_bounds__(128)
k_sage_gemm_wmma(const unsigned short* __restrict__ apack_u16,
                 const unsigned short* __restrict__ bpack_u16,
                 const float* __restrict__ bl,
                 float* __restrict__ out, int Nd, int combine) {
  __shared__ __align__(16) unsigned short smemB[16 * 256];     // 8KB weight tile

  int tid  = threadIdx.x;
  int wave = tid >> 5;
  int lane = tid & 31;
  int nt   = blockIdx.y;                                // 16-col tile (0..7)

  // --- TDM: stage Bpack[nt*16 .. nt*16+15][0:256] into LDS -----------------
  if (tid == 0) {
    unsigned long long ga =
        (unsigned long long)(size_t)(bpack_u16 + (size_t)nt * 16 * 256);
    unsigned lds_off = (unsigned)(size_t)(void*)smemB;  // generic addr low 32 = LDS offset
    u32x4 g0;
    g0[0] = 1u;                                         // count=1, user descriptor
    g0[1] = lds_off;                                    // lds_addr (bytes)
    g0[2] = (unsigned)(ga & 0xFFFFFFFFu);               // global_addr[31:0]
    g0[3] = (unsigned)((ga >> 32) & 0x1FFFFFFu)         // global_addr[56:32]
            | (2u << 30);                               // type=2 ("image")
    i32x8 g1;
    g1[0] = 1 << 16;                                    // data_size=1 (2 bytes)
    g1[1] = 256 << 16;                                  // tensor_dim0[15:0]=256
    g1[2] = (0) | (128 << 16);                          // dim0 hi=0, tensor_dim1=128
    g1[3] = (0) | (256 << 16);                          // dim1 hi=0, tile_dim0=256
    g1[4] = 16;                                         // tile_dim1=16, tile_dim2=0
    g1[5] = 256;                                        // tensor_dim0_stride[31:0]
    g1[6] = 0;                                          // stride hi, dim1_stride lo
    g1[7] = 0;
    i32x4 gz4 = {0, 0, 0, 0};
    i32x8 gz8 = {0, 0, 0, 0, 0, 0, 0, 0};
    __builtin_amdgcn_tensor_load_to_lds(g0, g1, gz4, gz4, gz8, 0);  // TENSORcnt++
    __builtin_amdgcn_s_wait_tensorcnt(0);
  }
  __syncthreads();                                      // LDS tile visible to all waves

  int rowtiles = (Nd + 15) >> 4;
  int mt = blockIdx.x * 4 + wave;                       // 16-row tile index
  if (mt >= rowtiles) return;                           // wave-uniform exit

  int laneHalf = lane >> 4;
  int lm       = lane & 15;

  int row = mt * 16 + lm;                               // A row held by this lane
  if (row >= Nd) row = Nd - 1;                          // clamp: never stored
  int ncol = nt * 16 + lm;                              // B column held by this lane

  const __bf16* ap = (const __bf16*)apack_u16 + (size_t)row * 256;
  const __bf16* bp = (const __bf16*)smemB + (size_t)lm * 256;  // LDS row = local col

  v8f acc;
  float bias = bl[ncol];
#pragma unroll
  for (int r = 0; r < 8; ++r) acc[r] = bias;            // bl broadcast over rows

#pragma unroll
  for (int ks = 0; ks < 8; ++ks) {                      // K = 256 total
    int kb = ks * 32 + laneHalf * 8;
    union { v16bf v; uint4 q[2]; } A, B;
    A.q[0] = *(const uint4*)(ap + kb);                  // global_load_b128
    A.q[1] = *(const uint4*)(ap + kb + 16);
    B.q[0] = *(const uint4*)(bp + kb);                  // ds_load_b128
    B.q[1] = *(const uint4*)(bp + kb + 16);
    // D = A x B + C   (8 args: neg_a, A, neg_b, B, c_mod, C, reuse_a, reuse_b)
    acc = __builtin_amdgcn_wmma_f32_16x16x32_bf16(false, A.v, false, B.v,
                                                  (short)0, acc, false, false);
  }

  // C/D layout: VGPR r -> M = r + 8*laneHalf, N = lane%16
#pragma unroll
  for (int r = 0; r < 8; ++r) {
    int orow = mt * 16 + r + laneHalf * 8;
    if (orow < Nd) {
      size_t idx = (size_t)orow * HID + ncol;
      float v = acc[r];
      out[idx] = combine ? fmaxf(out[idx], v) : v;
    }
  }
}

// ------------------------------ output head --------------------------------

// out[n, j] = sum_k relu(h[n,k]) * W[j,k] + b[j]      (128 -> 64)
__global__ void k_final_linear(const float* __restrict__ h, const float* __restrict__ W,
                               const float* __restrict__ bias, float* __restrict__ out,
                               int nrows) {
  long long t = (long long)blockIdx.x * blockDim.x + threadIdx.x;
  if (t >= (long long)nrows * OUTF) return;
  int n = (int)(t / OUTF), j = (int)(t % OUTF);
  const float* hr = h + (size_t)n * HID;
  const float* wr = W + (size_t)j * HID;
  float s = bias[j];
  for (int k = 0; k < HID; ++k) s += fmaxf(hr[k], 0.0f) * wr[k];
  out[(size_t)n * OUTF + j] = s;
}

// ------------------------------ host orchestration -------------------------

// Node types: 0=Patient 1=Admission 2=Diagnosis 3=Medication 4=Procedure 5=LabTest
static const int  kRows[6]   = {50000, 100000, 20000, 10000, 15000, 5000};
static const int  kRowOff[6] = {0, 50000, 150000, 170000, 180000, 195000};
static const long long kOutOff[6] = {0LL, 3200000LL, 9600000LL, 10880000LL,
                                     11520000LL, 12480000LL};
// relations in EDGE_TYPES order
static const int kRelSrc[10]  = {0, 1, 1, 2, 1, 3, 1, 4, 1, 5};
static const int kRelDst[10]  = {1, 0, 2, 1, 3, 1, 4, 1, 5, 1};
static const int kRelMode[10] = {0, 0, 0, 1, 0, 1, 0, 1, 0, 1};  // 0=write, 1=max

// input indices (setup_inputs dict insertion order, flattened depth-first)
#define IN_XP        0
#define IN_XA        1
#define IN_EDGE0     2    // 2..11, one [2,E] per relation
#define IN_PAT_W     12
#define IN_PAT_B     13
#define IN_ADM_W     14
#define IN_ADM_B     15
#define IN_PAT_BN_G  16
#define IN_PAT_BN_B  17
#define IN_ADM_BN_G  18
#define IN_ADM_BN_B  19
#define IN_EMB0      20   // diag, med, proc, lab = 20..23
#define IN_CONV1     24   // 24 + 30*layer + 3*rel : Wl, bl, Wr
#define IN_BN1       114  // 114 + 2*type : g, b
#define IN_BN2       126
#define IN_LIN       138  // 138 + 2*type : W, b

extern "C" void kernel_launch(void* const* d_in, const int* in_sizes, int n_in,
                              void* d_out, int out_size, void* d_ws, size_t ws_size,
                              hipStream_t stream) {
  const int TOTAL_ROWS = 200000;
  float* hA    = (float*)d_ws;                                  // 200000*128 f32
  float* hB    = hA   + (size_t)TOTAL_ROWS * HID;               // 200000*128 f32
  float* msum  = hB   + (size_t)TOTAL_ROWS * HID;               // 100000*128 f32
  float* cnt   = msum + (size_t)100000 * HID;                   // 100000 f32
  float* stats = cnt  + 100000;                                 // 256 f32
  unsigned short* apack = (unsigned short*)(stats + 256);       // 100000*256 bf16
  unsigned short* bpack = apack + (size_t)100000 * 256;         // 128*256 bf16

  auto F = [&](int i) { return (const float*)d_in[i]; };
  auto blocks = [](long long n, int bs) { return (unsigned)((n + bs - 1) / bs); };

  // --- encoder ---
  {
    long long nP = (long long)kRows[0] * HID;
    k_enc_linear<<<blocks(nP, 256), 256, 0, stream>>>(
        F(IN_XP), F(IN_PAT_W), F(IN_PAT_B), hA + (size_t)kRowOff[0] * HID,
        kRows[0], 32);
    k_zero<<<1, 256, 0, stream>>>(stats, 256);
    k_bn_stats<<<512, 256, 0, stream>>>(hA + (size_t)kRowOff[0] * HID, stats, kRows[0]);
    k_bn_apply<<<blocks(nP, 256), 256, 0, stream>>>(
        hA + (size_t)kRowOff[0] * HID, stats, F(IN_PAT_BN_G), F(IN_PAT_BN_B),
        kRows[0], 0);

    long long nA = (long long)kRows[1] * HID;
    k_enc_linear<<<blocks(nA, 256), 256, 0, stream>>>(
        F(IN_XA), F(IN_ADM_W), F(IN_ADM_B), hA + (size_t)kRowOff[1] * HID,
        kRows[1], 48);
    k_zero<<<1, 256, 0, stream>>>(stats, 256);
    k_bn_stats<<<512, 256, 0, stream>>>(hA + (size_t)kRowOff[1] * HID, stats, kRows[1]);
    k_bn_apply<<<blocks(nA, 256), 256, 0, stream>>>(
        hA + (size_t)kRowOff[1] * HID, stats, F(IN_ADM_BN_G), F(IN_ADM_BN_B),
        kRows[1], 0);

    for (int t = 2; t < 6; ++t) {                               // embedding types
      long long n = (long long)kRows[t] * HID;
      k_copy<<<blocks(n, 256), 256, 0, stream>>>(
          F(IN_EMB0 + (t - 2)), hA + (size_t)kRowOff[t] * HID, n);
    }
  }

  // --- three hetero-SAGE layers ---
  float* cur = hA;
  float* nxt = hB;
  for (int layer = 0; layer < 3; ++layer) {
    int convBase = IN_CONV1 + 30 * layer;
    for (int ri = 0; ri < 10; ++ri) {
      int ein = IN_EDGE0 + ri;
      int E = in_sizes[ein] / 2;
      const int* src = (const int*)d_in[ein];
      const int* dst = src + E;
      int st = kRelSrc[ri], dt = kRelDst[ri];
      int Nd = kRows[dt];

      // segment sum + count
      k_zero<<<blocks((long long)Nd * HID, 256), 256, 0, stream>>>(
          msum, (long long)Nd * HID);
      k_zero<<<blocks(Nd, 256), 256, 0, stream>>>(cnt, Nd);
      k_scatter_add<<<blocks((long long)E * 32, 256), 256, 0, stream>>>(
          cur + (size_t)kRowOff[st] * HID, src, dst, msum, cnt, E);

      // pack bf16 operands: A = [mean | h_dst], B = [Wl | Wr]
      k_pack_a<<<blocks((long long)Nd * 256, 256), 256, 0, stream>>>(
          msum, cnt, cur + (size_t)kRowOff[dt] * HID, apack, Nd);
      k_pack_w<<<blocks(HID * 256, 256), 256, 0, stream>>>(
          F(convBase + 3 * ri + 0), F(convBase + 3 * ri + 2), bpack);

      // WMMA GEMM + bias + (write | max-combine)
      int rowtiles = (Nd + 15) >> 4;
      dim3 grid((rowtiles + 3) / 4, HID / 16);
      k_sage_gemm_wmma<<<grid, 128, 0, stream>>>(
          apack, bpack, F(convBase + 3 * ri + 1),
          nxt + (size_t)kRowOff[dt] * HID, Nd, kRelMode[ri]);
    }
    if (layer < 2) {
      int bnBase = (layer == 0) ? IN_BN1 : IN_BN2;
      for (int t = 0; t < 6; ++t) {
        long long n = (long long)kRows[t] * HID;
        k_zero<<<1, 256, 0, stream>>>(stats, 256);
        k_bn_stats<<<512, 256, 0, stream>>>(nxt + (size_t)kRowOff[t] * HID, stats,
                                            kRows[t]);
        k_bn_apply<<<blocks(n, 256), 256, 0, stream>>>(
            nxt + (size_t)kRowOff[t] * HID, stats, F(bnBase + 2 * t),
            F(bnBase + 2 * t + 1), kRows[t], 1);
      }
    }
    float* tmp = cur; cur = nxt; nxt = tmp;
  }

  // --- per-type output head (ReLU fused) ---
  for (int t = 0; t < 6; ++t) {
    long long n = (long long)kRows[t] * OUTF;
    k_final_linear<<<blocks(n, 256), 256, 0, stream>>>(
        cur + (size_t)kRowOff[t] * HID, F(IN_LIN + 2 * t), F(IN_LIN + 2 * t + 1),
        (float*)d_out + kOutOff[t], kRows[t]);
  }
}
